// MambaAnglePredictor_40819369181223
// MI455X (gfx1250) — matrix operand order repaired and checked by reading, not compiled, not run on hardware
//
#include <hip/hip_runtime.h>
#include <hip/hip_bf16.h>
#include <stdint.h>

// ---------------- problem dimensions ----------------
#define B_       16
#define T_       4096
#define MROWS    (B_ * T_)      // 65536 token rows
#define DMODEL   192
#define DSTATE   64
#define HEADDIM_ 48
#define DCONV    4
#define DINNER   384            // EXPAND * D_MODEL
#define NHEADS_  8
#define CONVDIM  512            // D_INNER + 2*D_STATE
#define DINPROJ  904            // 2*D_INNER + 2*D_STATE + NHEADS
#define MIXDIM   32
#define INPUTDIM 4
#define MICDIM   12

typedef _Float16 f16;
typedef __attribute__((ext_vector_type(16))) _Float16 v16h;
typedef __attribute__((ext_vector_type(8)))  _Float16 v8h;
typedef __attribute__((ext_vector_type(8)))  float    v8f;

union FragH { v8h h[2]; v16h v; };

// N padded to the 64-column wave footprint so B-frag loads need no guard.
__host__ __device__ inline int npad_of(int N) {
  const int NT = (N + 15) >> 4;
  const int NG = (NT + 3) >> 2;
  return NG * 64;
}

// ---------------------------------------------------------------------------
// WMMA GEMM:  C[M,N] (f32) = A[M,K] (f16, row-major) * Bt[NPAD,K]^T (f16)
// One wave -> 16-row A tile x 4 N-tiles, K-loop step 32, software-pipelined
// (double-buffered fragments) so next k-step loads overlap current wmmas.
// K multiple of 32; M multiple of 16; Bt zero-padded to NPAD = NG*64 rows.
// A-frag per ISA layout: lanes 0-15 row=lane, K = k0+{0..7,16..23};
// lanes 16-31 same rows, K = k0+{8..15,24..31}.
// B-frag: lane n-col = lane&15; same K split as A-frag.
// ---------------------------------------------------------------------------
__global__ __launch_bounds__(256)
void gemm_wmma_f16(const f16* __restrict__ A, const f16* __restrict__ Bt,
                   float* __restrict__ C, int M, int N, int K) {
  const int NT = (N + 15) >> 4;
  const int NG = (NT + 3) >> 2;
  const int MT = M >> 4;
  const int wave = (blockIdx.x * blockDim.x + threadIdx.x) >> 5;
  if (wave >= MT * NG) return;
  const int lane = threadIdx.x & 31;
  const int mt = wave / NG;
  const int ng = wave - mt * NG;
  const int lr = lane & 15;
  const int hi = lane >> 4;

  v8f acc[4] = {};

  const f16* arow = A + (size_t)(mt * 16 + lr) * K + (hi ? 8 : 0);
  const f16* brow[4];
#pragma unroll
  for (int j = 0; j < 4; ++j) {
    const int nc = (((ng << 2) + j) << 4) + lr;   // always < NPAD
    brow[j] = Bt + (size_t)nc * K + (hi ? 8 : 0);
  }

  // prologue: fetch k-step 0
  FragH a_c, b_c[4];
  a_c.h[0] = *(const v8h*)(arow);
  a_c.h[1] = *(const v8h*)(arow + 16);
#pragma unroll
  for (int j = 0; j < 4; ++j) {
    b_c[j].h[0] = *(const v8h*)(brow[j]);
    b_c[j].h[1] = *(const v8h*)(brow[j] + 16);
  }

  // steady state: prefetch k0 while computing k0-32
  for (int k0 = 32; k0 < K; k0 += 32) {
    FragH a_n, b_n[4];
    a_n.h[0] = *(const v8h*)(arow + k0);
    a_n.h[1] = *(const v8h*)(arow + k0 + 16);
#pragma unroll
    for (int j = 0; j < 4; ++j) {
      b_n[j].h[0] = *(const v8h*)(brow[j] + k0);
      b_n[j].h[1] = *(const v8h*)(brow[j] + k0 + 16);
    }
#pragma unroll
    for (int j = 0; j < 4; ++j)
      acc[j] = __builtin_amdgcn_wmma_f32_16x16x32_f16(
          false, a_c.v, false, b_c[j].v, (short)0, acc[j], false, false);
    a_c = a_n;
#pragma unroll
    for (int j = 0; j < 4; ++j) b_c[j] = b_n[j];
  }

  // epilogue: last k-step
#pragma unroll
  for (int j = 0; j < 4; ++j)
    acc[j] = __builtin_amdgcn_wmma_f32_16x16x32_f16(
        false, a_c.v, false, b_c[j].v, (short)0, acc[j], false, false);

  const int rbase = mt * 16 + (hi ? 8 : 0);
#pragma unroll
  for (int j = 0; j < 4; ++j) {
    const int nc = (((ng << 2) + j) << 4) + lr;
    if (nc >= N) continue;                        // edge guard only on store
#pragma unroll
    for (int r = 0; r < 8; ++r)
      C[(size_t)(rbase + r) * N + nc] = acc[j][r];
  }
}

// ---- per-batch bias: b_in + relu(mic_pos @ w_mic + b_mic) -----------------
__global__ void mic_bias_kernel(const float* __restrict__ mic_pos,
                                const float* __restrict__ w_mic,
                                const float* __restrict__ b_mic,
                                const float* __restrict__ b_in,
                                float* __restrict__ bias) {
  const int b = blockIdx.x, d = threadIdx.x;
  float acc = b_mic[d];
  for (int k = 0; k < MICDIM; ++k)
    acc += mic_pos[b * MICDIM + k] * w_mic[k * DMODEL + d];
  bias[b * DMODEL + d] = b_in[d] + fmaxf(acc, 0.f);
}

// ---- f32 [K,N] -> f16 [NPAD,K] transpose with zero padding ----------------
__global__ void transpose_pad_f16_kernel(const float* __restrict__ src,
                                         f16* __restrict__ dst,
                                         int K, int N, int NPAD) {
  const int e = blockIdx.x * blockDim.x + threadIdx.x;   // over NPAD*K
  if (e >= NPAD * K) return;
  const int n = e / K, k = e - n * K;
  dst[e] = (n < N) ? (f16)src[(size_t)k * N + n] : (f16)0.f;
}

// ---- h = x @ w_in + bias[b], cast to f16 ----------------------------------
__global__ void embed_kernel(const float* __restrict__ x,
                             const float* __restrict__ w_in,
                             const float* __restrict__ bias,
                             f16* __restrict__ hbf) {
  const int e = blockIdx.x * blockDim.x + threadIdx.x;
  if (e >= MROWS * DMODEL) return;
  const int m = e / DMODEL, d = e - m * DMODEL;
  const int b = m >> 12;                      // T_ = 4096 rows per batch
  float acc = bias[b * DMODEL + d];
#pragma unroll
  for (int k = 0; k < INPUTDIM; ++k)
    acc += x[(size_t)m * INPUTDIM + k] * w_in[k * DMODEL + d];
  hbf[e] = (f16)acc;
}

// ---- depthwise causal conv (D_CONV=4) + SiLU over xBC slice ---------------
__global__ void conv_silu_kernel(const float* __restrict__ zx,
                                 const float* __restrict__ conv_w,
                                 const float* __restrict__ conv_b,
                                 float* __restrict__ xbc) {
  const int e = blockIdx.x * blockDim.x + threadIdx.x;
  if (e >= MROWS * CONVDIM) return;
  const int m = e >> 9, c = e & (CONVDIM - 1);
  const int t = m & (T_ - 1);
  float acc = conv_b[c];
#pragma unroll
  for (int k = 0; k < DCONV; ++k) {
    const int tt = t - (DCONV - 1) + k;
    if (tt >= 0)
      acc += zx[(size_t)(m - (DCONV - 1) + k) * DINPROJ + DINNER + c] *
             conv_w[c * DCONV + k];
  }
  xbc[e] = acc / (1.f + __expf(-acc));        // SiLU
}

// ---- dt = softplus(raw + dt_bias) -----------------------------------------
__global__ void dt_kernel(const float* __restrict__ zx,
                          const float* __restrict__ dt_bias,
                          float* __restrict__ dt) {
  const int e = blockIdx.x * blockDim.x + threadIdx.x;
  if (e >= MROWS * NHEADS_) return;
  const int m = e >> 3, hh = e & 7;
  const float r = zx[(size_t)m * DINPROJ + DINNER + CONVDIM + hh] + dt_bias[hh];
  dt[e] = (r > 20.f) ? r : log1pf(__expf(r));
}

// ---- selective scan: one block per (batch, head); lane p holds h[p, 0..63]
// in 64 VGPRs. B/C/x/dt staged through LDS in chunks of 8 timesteps. --------
__global__ __launch_bounds__(64)
void scan_kernel(const float* __restrict__ xbc, const float* __restrict__ dt,
                 const float* __restrict__ A_log, const float* __restrict__ D_skip,
                 float* __restrict__ y) {
  const int bh = blockIdx.x;
  const int b = bh >> 3, head = bh & 7;
  const int tid = threadIdx.x;
  const float Aneg = -__expf(A_log[head]);
  const float Dh = D_skip[head];

  __shared__ float sB[8][DSTATE];
  __shared__ float sC[8][DSTATE];
  __shared__ float sX[8][HEADDIM_];
  __shared__ float sDt[8];

  float h[DSTATE];
#pragma unroll
  for (int n = 0; n < DSTATE; ++n) h[n] = 0.f;

  const size_t rowbase = (size_t)b * T_;
  for (int t0 = 0; t0 < T_; t0 += 8) {
    for (int i = tid; i < 8 * DSTATE; i += 64) {
      const int c = i >> 6, n = i & 63;
      const size_t base = (rowbase + t0 + c) * (size_t)CONVDIM;
      sB[c][n] = xbc[base + DINNER + n];
      sC[c][n] = xbc[base + DINNER + DSTATE + n];
    }
    for (int i = tid; i < 8 * HEADDIM_; i += 64) {
      const int c = i / HEADDIM_, p = i - c * HEADDIM_;
      sX[c][p] = xbc[(rowbase + t0 + c) * (size_t)CONVDIM + head * HEADDIM_ + p];
    }
    if (tid < 8) sDt[tid] = dt[(rowbase + t0 + tid) * NHEADS_ + head];
    __syncthreads();

    if (tid < HEADDIM_) {
      const int p = tid;
      for (int c = 0; c < 8; ++c) {
        const float d = sDt[c];
        const float decay = __expf(d * Aneg);
        const float xv = sX[c][p];
        const float scale = d * xv;
        float yv = 0.f;
#pragma unroll
        for (int n = 0; n < DSTATE; ++n) {
          h[n] = decay * h[n] + scale * sB[c][n];
          yv += h[n] * sC[c][n];
        }
        y[(rowbase + t0 + c) * (size_t)DINNER + head * HEADDIM_ + p] = yv + Dh * xv;
      }
    }
    __syncthreads();
  }
}

// ---- y *= silu(z); RMSNorm; * norm_w; -> f16 ------------------------------
__global__ __launch_bounds__(128)
void norm_gate_kernel(const float* __restrict__ ys, const float* __restrict__ zx,
                      const float* __restrict__ norm_w, f16* __restrict__ ybf) {
  const int m = blockIdx.x;
  const int tid = threadIdx.x;                // 128 threads, 3 cols each
  float v[3];
  float ss = 0.f;
#pragma unroll
  for (int jj = 0; jj < 3; ++jj) {
    const int j = tid + jj * 128;
    const float yv = ys[(size_t)m * DINNER + j];
    const float z = zx[(size_t)m * DINPROJ + j];
    const float g = yv * (z / (1.f + __expf(-z)));
    v[jj] = g;
    ss += g * g;
  }
  __shared__ float red[128];
  red[tid] = ss;
  __syncthreads();
  for (int s = 64; s > 0; s >>= 1) {
    if (tid < s) red[tid] += red[tid + s];
    __syncthreads();
  }
  const float rstd = rsqrtf(red[0] / (float)DINNER + 1e-5f);
#pragma unroll
  for (int jj = 0; jj < 3; ++jj) {
    const int j = tid + jj * 128;
    ybf[(size_t)m * DINNER + j] = (f16)(v[jj] * rstd * norm_w[j]);
  }
}

__global__ void zero_kernel(float* __restrict__ p, int n) {
  const int i = blockIdx.x * blockDim.x + threadIdx.x;
  if (i < n) p[i] = 0.f;
}

// ---- mix MLP + mean over T (8 rows / block, weights in LDS) ---------------
__global__ __launch_bounds__(256)
void mix_kernel(const float* __restrict__ hout,
                const float* __restrict__ w1, const float* __restrict__ b1,
                const float* __restrict__ w2, const float* __restrict__ b2,
                float* __restrict__ accum) {
  __shared__ float w1s[DMODEL * MIXDIM];
  __shared__ float w2s[MIXDIM * MIXDIM];
  __shared__ float hs[8][DMODEL];
  __shared__ float m1s[8][MIXDIM];
  __shared__ float accs[MIXDIM];
  const int tid = threadIdx.x;
  const int r0 = blockIdx.x * 8;
  for (int i = tid; i < DMODEL * MIXDIM; i += 256) w1s[i] = w1[i];
  for (int i = tid; i < MIXDIM * MIXDIM; i += 256) w2s[i] = w2[i];
  for (int i = tid; i < 8 * DMODEL; i += 256) {
    const int r = i / DMODEL, k = i - r * DMODEL;
    hs[r][k] = hout[(size_t)(r0 + r) * DMODEL + k];
  }
  if (tid < MIXDIM) accs[tid] = 0.f;
  __syncthreads();

  const int r = tid >> 5, j = tid & 31;
  float a1 = b1[j];
  for (int k = 0; k < DMODEL; ++k) a1 += hs[r][k] * w1s[k * MIXDIM + j];
  m1s[r][j] = fmaxf(a1, 0.f);
  __syncthreads();

  float a2 = b2[j];
#pragma unroll
  for (int k = 0; k < MIXDIM; ++k) a2 += m1s[r][k] * w2s[k * MIXDIM + j];
  atomicAdd(&accs[j], fmaxf(a2, 0.f));
  __syncthreads();
  if (tid < MIXDIM) {
    const int batch = r0 >> 12;               // 4096 rows per batch
    atomicAdd(&accum[batch * MIXDIM + tid], accs[tid]);
  }
}

__global__ void final_kernel(const float* __restrict__ accum,
                             const float* __restrict__ w_out,
                             const float* __restrict__ b_out,
                             float* __restrict__ out) {
  const int b = threadIdx.x;
  if (b < B_) {
    float s = 0.f;
    for (int j = 0; j < MIXDIM; ++j) s += accum[b * MIXDIM + j] * w_out[j];
    out[b] = s * (1.f / (float)T_) + b_out[0];
  }
}

// ---------------------------------------------------------------------------
// Workspace need: ~550 MB (zxbcdt f32 dominates; out-proj result aliases it).
// ---------------------------------------------------------------------------
extern "C" void kernel_launch(void* const* d_in, const int* in_sizes, int n_in,
                              void* d_out, int out_size, void* d_ws, size_t ws_size,
                              hipStream_t stream) {
  (void)in_sizes; (void)n_in; (void)out_size; (void)ws_size;
  const float* x        = (const float*)d_in[0];
  const float* mic_pos  = (const float*)d_in[1];
  const float* w_in     = (const float*)d_in[2];
  const float* b_in     = (const float*)d_in[3];
  const float* w_mic    = (const float*)d_in[4];
  const float* b_mic    = (const float*)d_in[5];
  const float* w_inproj = (const float*)d_in[6];
  const float* conv_w   = (const float*)d_in[7];
  const float* conv_b   = (const float*)d_in[8];
  const float* dt_bias  = (const float*)d_in[9];
  const float* A_log    = (const float*)d_in[10];
  const float* D_skip   = (const float*)d_in[11];
  const float* norm_w   = (const float*)d_in[12];
  const float* w_outproj= (const float*)d_in[13];
  const float* w_mix1   = (const float*)d_in[14];
  const float* b_mix1   = (const float*)d_in[15];
  const float* w_mix2   = (const float*)d_in[16];
  const float* b_mix2   = (const float*)d_in[17];
  const float* w_out    = (const float*)d_in[18];
  const float* b_out    = (const float*)d_in[19];
  float* out = (float*)d_out;

  const int NPAD1 = npad_of(DINPROJ);   // 960
  const int NPAD2 = npad_of(DMODEL);    // 192 (already aligned)

  char* ws = (char*)d_ws;
  size_t off = 0;
  auto take = [&](size_t bytes) {
    size_t o = off;
    off = (off + bytes + 255) & ~(size_t)255;
    return o;
  };
  float* bias = (float*)(ws + take((size_t)B_ * DMODEL * 4));
  f16*   hbf  = (f16*)  (ws + take((size_t)MROWS * DMODEL * 2));
  f16*   wA_t = (f16*)  (ws + take((size_t)NPAD1 * DMODEL * 2));
  f16*   wO_t = (f16*)  (ws + take((size_t)NPAD2 * DINNER * 2));
  float* zx   = (float*)(ws + take((size_t)MROWS * DINPROJ * 4));
  float* xbc  = (float*)(ws + take((size_t)MROWS * CONVDIM * 4));
  float* dt   = (float*)(ws + take((size_t)MROWS * NHEADS_ * 4));
  float* ys   = (float*)(ws + take((size_t)MROWS * DINNER * 4));
  f16*   ybf  = (f16*)  (ws + take((size_t)MROWS * DINNER * 2));
  float* accum= (float*)(ws + take((size_t)B_ * MIXDIM * 4));
  float* hout = zx;  // zxbcdt is dead after norm_gate -> reuse for out-proj

  mic_bias_kernel<<<B_, DMODEL, 0, stream>>>(mic_pos, w_mic, b_mic, b_in, bias);
  transpose_pad_f16_kernel<<<(NPAD1 * DMODEL + 255) / 256, 256, 0, stream>>>(
      w_inproj, wA_t, DMODEL, DINPROJ, NPAD1);
  transpose_pad_f16_kernel<<<(NPAD2 * DINNER + 255) / 256, 256, 0, stream>>>(
      w_outproj, wO_t, DINNER, DMODEL, NPAD2);
  embed_kernel<<<(MROWS * DMODEL + 255) / 256, 256, 0, stream>>>(x, w_in, bias, hbf);

  {  // GEMM1: zxbcdt = h @ w_inproj  (M=65536, K=192, N=904)
    const int MT = MROWS / 16, NT = (DINPROJ + 15) / 16, NG = (NT + 3) / 4;
    const int blocks = (MT * NG * 32 + 255) / 256;
    gemm_wmma_f16<<<blocks, 256, 0, stream>>>(hbf, wA_t, zx, MROWS, DINPROJ, DMODEL);
  }

  conv_silu_kernel<<<(MROWS * CONVDIM + 255) / 256, 256, 0, stream>>>(
      zx, conv_w, conv_b, xbc);
  dt_kernel<<<(MROWS * NHEADS_ + 255) / 256, 256, 0, stream>>>(zx, dt_bias, dt);

  scan_kernel<<<B_ * NHEADS_, 64, 0, stream>>>(xbc, dt, A_log, D_skip, ys);

  norm_gate_kernel<<<MROWS, 128, 0, stream>>>(ys, zx, norm_w, ybf);

  {  // GEMM2: hout = y @ w_outproj  (M=65536, K=384, N=192)
    const int MT = MROWS / 16, NT = (DMODEL + 15) / 16, NG = (NT + 3) / 4;
    const int blocks = (MT * NG * 32 + 255) / 256;
    gemm_wmma_f16<<<blocks, 256, 0, stream>>>(ybf, wO_t, hout, MROWS, DMODEL, DINNER);
  }

  zero_kernel<<<(B_ * MIXDIM + 255) / 256, 256, 0, stream>>>(accum, B_ * MIXDIM);
  mix_kernel<<<MROWS / 8, 256, 0, stream>>>(hout, w_mix1, b_mix1, w_mix2, b_mix2, accum);
  final_kernel<<<1, 32, 0, stream>>>(accum, w_out, b_out, out);
}
